// TorchAttention_20074677141655
// MI455X (gfx1250) — matrix operand order repaired and checked by reading, not compiled, not run on hardware
//
#include <hip/hip_runtime.h>

// ---------------------------------------------------------------------------
// ALiBi causal attention block for MI455X (gfx1250, wave32, WMMA bf16).
// B=2, S=2048, H=16, Dh=128, Dmodel=2048.
// Pipeline: cvt fp32->bf16, GEMM(Q/K/V), flash-attn, GEMM(Wo -> fp32 out).
// GEMM uses double-buffered LDS with global_load_async_to_lds_b128 staging.
// ---------------------------------------------------------------------------

#define BATCH  2
#define SEQ    2048
#define NHEAD  16
#define DH     128
#define DMODEL 2048

typedef __attribute__((ext_vector_type(16))) __bf16 v16bf;
typedef __attribute__((ext_vector_type(8)))  float  v8f;
typedef __attribute__((ext_vector_type(4)))  unsigned int u32x4;

__device__ __forceinline__ v8f wmma_bf16(v16bf a, v16bf b, v8f c) {
  // D = A(16x32 bf16) * B(32x16 bf16) + C(16x16 f32)
  return __builtin_amdgcn_wmma_f32_16x16x32_bf16(false, a, false, b,
                                                 (short)0, c, false, false);
}

// A-fragment: elements 0..7 from p0 (K=k0..k0+7), 8..15 from p1 (K=k0+16..k0+23)
__device__ __forceinline__ v16bf ld_frag_split(const __bf16* p0, const __bf16* p1) {
  union { v16bf v; u32x4 u[2]; } r;
  r.u[0] = *(const u32x4*)p0;
  r.u[1] = *(const u32x4*)p1;
  return r.v;
}
// B-fragment: elements 0..7 from p (K=k0..k0+7), 8..15 from p+16 (K=k0+16..k0+23)
__device__ __forceinline__ v16bf ld_frag16(const __bf16* p) {
  return ld_frag_split(p, p + 16);
}

// gfx1250 async copy: 16 bytes global -> LDS, tracked by ASYNCcnt.
// LDS operand is the 32-bit LDS byte offset (low 32 bits of the generic addr).
__device__ __forceinline__ void async_b128(void* lds, const void* g) {
  unsigned loff = (unsigned)(size_t)lds;
  asm volatile("global_load_async_to_lds_b128 %0, %1, off"
               :: "v"(loff), "v"((unsigned long long)(size_t)g) : "memory");
}
__device__ __forceinline__ void wait_async() {
  asm volatile("s_wait_asynccnt 0x0" ::: "memory");
}

// ---------------------------------------------------------------------------
// fp32 -> bf16 elementwise conversion
// ---------------------------------------------------------------------------
__global__ __launch_bounds__(256) void cvt_bf16(const float* __restrict__ in,
                                                __bf16* __restrict__ out, int n) {
  int i = blockIdx.x * 256 + threadIdx.x;
  int stride = gridDim.x * 256;
  for (; i < n; i += stride) out[i] = (__bf16)in[i];
}

// ---------------------------------------------------------------------------
// bf16 GEMM: C[M,N] = A[M,K] * B[K,N].  M mult of 256, N mult of 128, K of 32.
// Block: 256 threads (8 waves), 256x128 tile, K-step 32, double-buffered LDS.
// Wave (wm,wn) in 4x2 grid computes 64x64 = 4x4 WMMA subtiles.
// A tile staged with async-to-LDS copies; B tile transposed through VGPRs.
// ---------------------------------------------------------------------------
template <bool OUT_BF16>
__global__ __launch_bounds__(256) void gemm_bf16_k(const __bf16* __restrict__ A,
                                                   const __bf16* __restrict__ B,
                                                   void* __restrict__ Cp,
                                                   int M, int N, int K) {
  __shared__ __bf16 lA[2][256 * 32];  // [m][k]           (2 x 16 KB)
  __shared__ __bf16 lB[2][128 * 32];  // [n][k] transposed (2 x 8 KB)
  const int tid = threadIdx.x;
  const int w   = tid >> 5;
  const int l   = tid & 31;
  const int hl  = l >> 4;     // half-wave select (ISA fragment split)
  const int ll  = l & 15;
  const int wm  = w >> 1;     // 0..3 : 64-row slab
  const int wn  = w & 1;      // 0..1 : 64-col slab
  const int m0  = blockIdx.y * 256;
  const int n0  = blockIdx.x * 128;

  v8f acc[4][4];
#pragma unroll
  for (int i = 0; i < 4; ++i)
#pragma unroll
    for (int j = 0; j < 4; ++j) acc[i][j] = {};

  const __bf16* arow = A + (size_t)(m0 + tid) * K;  // one A row per thread

  // ---- staging helpers (as lambdas so both prologue and loop share them) ---
  auto stageA = [&](int k0, int buf) {
    // 256x32 tile: each thread async-copies its row (64 B) as 4 x b128.
    __bf16* la = &lA[buf][tid * 32];
    const __bf16* ga = arow + k0;
#pragma unroll
    for (int c = 0; c < 4; ++c) async_b128(la + c * 8, ga + c * 8);
  };
  auto stageB = [&](int k0, int buf) {
    // 32x128 tile transposed into [n][k]; coalesced global reads along N.
#pragma unroll
    for (int i = 0; i < 16; ++i) {
      int lin = tid + i * 256;        // 0..4095
      int kk  = lin >> 7;             // 0..31
      int nn  = lin & 127;
      lB[buf][nn * 32 + kk] = B[(size_t)(k0 + kk) * N + n0 + nn];
    }
  };

  const int nk = K >> 5;
  stageA(0, 0);
  stageB(0, 0);
  wait_async();
  __syncthreads();

  for (int i = 0; i < nk; ++i) {
    const int pb = i & 1;
    if (i + 1 < nk) {                 // prefetch next K-slab into other buffer
      stageA((i + 1) << 5, pb ^ 1);
      stageB((i + 1) << 5, pb ^ 1);
    }

    v16bf af[4], bfr[4];
#pragma unroll
    for (int mt = 0; mt < 4; ++mt) {
      const __bf16* p = &lA[pb][(wm * 64 + mt * 16 + ll) * 32 + hl * 8];
      af[mt] = ld_frag_split(p, p + 16);
    }
#pragma unroll
    for (int nt = 0; nt < 4; ++nt) {
      const __bf16* p = &lB[pb][(wn * 64 + nt * 16 + ll) * 32 + hl * 8];
      bfr[nt] = ld_frag16(p);
    }
#pragma unroll
    for (int mt = 0; mt < 4; ++mt)
#pragma unroll
      for (int nt = 0; nt < 4; ++nt)
        acc[mt][nt] = wmma_bf16(af[mt], bfr[nt], acc[mt][nt]);

    wait_async();                     // async writes to buf^1 have landed
    __syncthreads();                  // everyone done reading buf, writing buf^1
  }

  // Epilogue: C-fragment layout -> row = vgpr + 8*(lane>=16), col = lane&15
#pragma unroll
  for (int mt = 0; mt < 4; ++mt)
#pragma unroll
    for (int nt = 0; nt < 4; ++nt)
#pragma unroll
      for (int v = 0; v < 8; ++v) {
        int gr = m0 + wm * 64 + mt * 16 + hl * 8 + v;
        int gc = n0 + wn * 64 + nt * 16 + ll;
        float val = acc[mt][nt][v];
        if (OUT_BF16) ((__bf16*)Cp)[(size_t)gr * N + gc] = (__bf16)val;
        else          ((float*)Cp)[(size_t)gr * N + gc]  = val;
      }
}

// ---------------------------------------------------------------------------
// Flash attention with causal mask + ALiBi.
// Grid: BATCH * NHEAD * (SEQ/128) blocks; 256 threads (8 waves).
// Each wave owns 16 query rows; block iterates key tiles of 32.
// Q,K,V,O layout: [b][s][h*DH + d] bf16.
// ---------------------------------------------------------------------------
__global__ __launch_bounds__(256) void attn_k(const __bf16* __restrict__ Q,
                                              const __bf16* __restrict__ K,
                                              const __bf16* __restrict__ V,
                                              __bf16* __restrict__ O) {
  __shared__ __bf16 lVt[DH * 32];       // [d][krel] transposed V tile (8 KB)
  __shared__ __bf16 lP[8][16 * 32];     // per-wave P transpose buffer (8 KB)

  const int tid = threadIdx.x;
  const int w   = tid >> 5;
  const int l   = tid & 31;
  const int hl  = l >> 4;
  const int ll  = l & 15;

  const int bx = blockIdx.x;
  const int qt = bx & 15;               // SEQ/128 = 16
  const int h  = (bx >> 4) & (NHEAD - 1);
  const int b  = bx >> 8;
  const int qblock = qt * 128;
  const int qbase  = qblock + w * 16;

  const float scale = 0.08838834764831845f;          // 1/sqrt(128)
  const float slope = exp2f(-0.5f * (float)(h + 1)); // 2^(-8(h+1)/16)

  // Q fragments (4 d-slices of 32), kept in registers for the whole kernel.
  v16bf qf[4];
  {
    const __bf16* qp = Q + ((size_t)b * SEQ + qbase + ll) * DMODEL + h * DH;
#pragma unroll
    for (int ds = 0; ds < 4; ++ds) {
      const __bf16* p = qp + ds * 32 + hl * 8;
      qf[ds] = ld_frag_split(p, p + 16);
    }
  }

  v8f o[8];
#pragma unroll
  for (int nt = 0; nt < 8; ++nt) o[nt] = {};
  float mrow[8], lrow[8];
#pragma unroll
  for (int v = 0; v < 8; ++v) { mrow[v] = -1e30f; lrow[v] = 0.0f; }

  const int kend = qblock + 128;   // causal bound for this 128-row block
  for (int kb = 0; kb < kend; kb += 32) {
    // Stage V tile transposed: lVt[d][krel]
#pragma unroll
    for (int i = 0; i < 16; ++i) {
      int lin = tid + i * 256;     // 0..4095
      int kk  = lin >> 7;          // 0..31
      int dd  = lin & 127;         // coalesced along d
      lVt[dd * 32 + kk] = V[((size_t)b * SEQ + kb + kk) * DMODEL + h * DH + dd];
    }
    __syncthreads();

    // Scores S = Q * K^T for 32 keys: d-slice outer so only 2 K-fragments
    // are live at a time (keeps the wave under the 256-VGPR window).
    v8f s[2];
    s[0] = {}; s[1] = {};
    {
      const __bf16* kp0 = K + ((size_t)b * SEQ + kb + ll) * DMODEL
                            + h * DH + hl * 8;
      const __bf16* kp1 = kp0 + (size_t)16 * DMODEL;
#pragma unroll
      for (int ds = 0; ds < 4; ++ds) {
        v16bf kf0 = ld_frag16(kp0 + ds * 32);
        v16bf kf1 = ld_frag16(kp1 + ds * 32);
        s[0] = wmma_bf16(qf[ds], kf0, s[0]);
        s[1] = wmma_bf16(qf[ds], kf1, s[1]);
      }
    }

    // ALiBi + causal mask + online softmax update.
    float pf[2][8];
#pragma unroll
    for (int v = 0; v < 8; ++v) {
      const int i = qbase + hl * 8 + v;
      float tmax = -1e30f;
#pragma unroll
      for (int jt = 0; jt < 2; ++jt) {
        const int j = kb + jt * 16 + ll;
        float val = s[jt][v] * scale + slope * (float)(j - i);
        if (j > i) val = -1e30f;
        s[jt][v] = val;
        tmax = fmaxf(tmax, val);
      }
#pragma unroll
      for (int x = 1; x < 16; x <<= 1)
        tmax = fmaxf(tmax, __shfl_xor(tmax, x, 32));
      const float mnew  = fmaxf(mrow[v], tmax);
      const float alpha = __expf(mrow[v] - mnew);
      mrow[v] = mnew;
      float rsum = 0.0f;
#pragma unroll
      for (int jt = 0; jt < 2; ++jt) {
        float p = __expf(s[jt][v] - mnew);
        pf[jt][v] = p;
        rsum += p;
      }
#pragma unroll
      for (int x = 1; x < 16; x <<= 1)
        rsum += __shfl_xor(rsum, x, 32);
      lrow[v] = lrow[v] * alpha + rsum;
#pragma unroll
      for (int nt = 0; nt < 8; ++nt) o[nt][v] *= alpha;
    }

    // Transpose P (C-layout) into A-fragment layout via per-wave LDS.
    __bf16* lp = &lP[w][0];
#pragma unroll
    for (int jt = 0; jt < 2; ++jt)
#pragma unroll
      for (int v = 0; v < 8; ++v)
        lp[(hl * 8 + v) * 32 + jt * 16 + ll] = (__bf16)pf[jt][v];
    // LDS ops within a wave are in-order -> safe to read back immediately.
    v16bf pa;
    {
      const __bf16* p = lp + ll * 32 + hl * 8;
      pa = ld_frag_split(p, p + 16);
    }

    // O += P(16x32) * V(32x128): 8 WMMAs against transposed V in LDS.
#pragma unroll
    for (int nt = 0; nt < 8; ++nt) {
      const __bf16* vp = lVt + (nt * 16 + ll) * 32 + hl * 8;
      o[nt] = wmma_bf16(pa, ld_frag16(vp), o[nt]);
    }
    __syncthreads();
  }

  // Epilogue: O / l, store bf16 into [b][s][h*DH + d].
#pragma unroll
  for (int nt = 0; nt < 8; ++nt)
#pragma unroll
    for (int v = 0; v < 8; ++v) {
      const int row = qbase + hl * 8 + v;
      const int col = h * DH + nt * 16 + ll;
      O[((size_t)b * SEQ + row) * DMODEL + col] = (__bf16)(o[nt][v] / lrow[v]);
    }
}

// ---------------------------------------------------------------------------
// Host-side launcher
// ---------------------------------------------------------------------------
extern "C" void kernel_launch(void* const* d_in, const int* in_sizes, int n_in,
                              void* d_out, int out_size, void* d_ws, size_t ws_size,
                              hipStream_t stream) {
  (void)in_sizes; (void)n_in; (void)out_size; (void)ws_size;

  const float* x  = (const float*)d_in[0];
  // d_in[1] = attention_mask: causal, applied analytically -> unused.
  const float* Wq = (const float*)d_in[2];
  const float* Wk = (const float*)d_in[3];
  const float* Wv = (const float*)d_in[4];
  const float* Wo = (const float*)d_in[5];

  const int xN = BATCH * SEQ * DMODEL;        // 8,388,608
  const int wN = DMODEL * DMODEL;             // 4,194,304
  const int M  = BATCH * SEQ;                 // 4096

  // Workspace layout (bf16 elements), ~112 MB total.
  __bf16* base = (__bf16*)d_ws;
  __bf16* xb  = base;
  __bf16* wqb = xb  + xN;
  __bf16* wkb = wqb + wN;
  __bf16* wvb = wkb + wN;
  __bf16* wob = wvb + wN;
  __bf16* Qb  = wob + wN;
  __bf16* Kb  = Qb  + xN;
  __bf16* Vb  = Kb  + xN;
  __bf16* Ab  = Vb  + xN;   // attention output (pre-Wo)

  // 1) fp32 -> bf16 conversions
  cvt_bf16<<<2048, 256, 0, stream>>>(x,  xb,  xN);
  cvt_bf16<<<2048, 256, 0, stream>>>(Wq, wqb, wN);
  cvt_bf16<<<2048, 256, 0, stream>>>(Wk, wkb, wN);
  cvt_bf16<<<2048, 256, 0, stream>>>(Wv, wvb, wN);
  cvt_bf16<<<2048, 256, 0, stream>>>(Wo, wob, wN);

  // 2) Q/K/V projections: [4096x2048] x [2048x2048] -> bf16
  dim3 ggrid(DMODEL / 128, M / 256);
  gemm_bf16_k<true><<<ggrid, 256, 0, stream>>>(xb, wqb, Qb, M, DMODEL, DMODEL);
  gemm_bf16_k<true><<<ggrid, 256, 0, stream>>>(xb, wkb, Kb, M, DMODEL, DMODEL);
  gemm_bf16_k<true><<<ggrid, 256, 0, stream>>>(xb, wvb, Vb, M, DMODEL, DMODEL);

  // 3) Flash attention (causal + ALiBi)
  attn_k<<<BATCH * NHEAD * (SEQ / 128), 256, 0, stream>>>(Qb, Kb, Vb, Ab);

  // 4) Output projection -> fp32 d_out
  gemm_bf16_k<false><<<ggrid, 256, 0, stream>>>(Ab, wob, d_out, M, DMODEL, DMODEL);
}
